// EST_module_18588618457700
// MI455X (gfx1250) — matrix operand order repaired and checked by reading, not compiled, not run on hardware
//
#include <hip/hip_runtime.h>
#include <hip/hip_bf16.h>

// ---------------------------------------------------------------------------
// EST module fused pipeline for gfx1250 (MI455X, wave32, WMMA bf16).
// Shapes: E=4096 tokens, S_SPH=49, S_GRID=128, C=128, NE=8, HID=512.
// One workgroup (8 waves) per token; all intermediates stay in LDS; all GEMMs
// use v_wmma_f32_16x16x32_bf16 with f32 accumulation.
// MoE refinements this round: gate folded into h (single accumulator set),
// xhat A-fragments hoisted across the whole expert loop.
// ---------------------------------------------------------------------------

typedef __bf16 bf16_t;
typedef __attribute__((ext_vector_type(16))) __bf16 v16bf;
typedef __attribute__((ext_vector_type(8)))  float  v8f;

#define E_TOK   4096
#define S_SPH   49
#define S_GRID  128
#define CH      128
#define NEXP    8
#define HID     512
#define LDF     129            // padded f32 LDS stride (bank-conflict-free rows)

__device__ __forceinline__ v8f wmma_bf16(v16bf a, v16bf b, v8f c) {
  return __builtin_amdgcn_wmma_f32_16x16x32_bf16(
      /*neg_a=*/false, a, /*neg_b=*/false, b,
      /*c_mod=*/(short)0, c, /*reuse_a=*/false, /*reuse_b=*/false);
}

// A fragment (16x32 bf16, M x K): lane row = m0+(lane&15);
// lo-half lanes hold K in [k0+0,8)∪[k0+16,24), hi-half [k0+8,16)∪[k0+24,32).
__device__ __forceinline__ v16bf fragA(const bf16_t* base, int ld, int row,
                                       int k0, int hi) {
  const bf16_t* p = base + row * ld + k0 + hi * 8;
  union { v16bf v; bf16_t e[16]; } f;
#pragma unroll
  for (int t = 0; t < 8; ++t) { f.e[t] = p[t]; f.e[t + 8] = p[t + 16]; }
  return f.v;
}

// B fragment (32x16 bf16, K x N) loaded from Bt (N-major, K contiguous):
// lane column = n0+(lane&15); lo-half lanes K in [k0+0,8)∪[k0+16,24), hi-half [k0+8,16)∪[k0+24,32).
__device__ __forceinline__ v16bf fragB(const bf16_t* baseT, int ld, int row,
                                       int k0, int hi) {
  const bf16_t* p = baseT + row * ld + k0 + hi * 8;
  union { v16bf v; bf16_t e[16]; } f;
#pragma unroll
  for (int t = 0; t < 8; ++t) { f.e[t] = p[t]; f.e[t + 8] = p[t + 16]; }
  return f.v;
}

__device__ __forceinline__ float silu_f(float x) {
  return x / (1.f + __expf(-x));
}

// ---------------------------------------------------------------------------
// Prep kernels: bf16 conversion / folding into workspace (run once per call).
// ---------------------------------------------------------------------------

__global__ void est_prep_small(const float* __restrict__ Ygf,
                               const float* __restrict__ Yinvf,
                               const float* __restrict__ wqf,
                               const float* __restrict__ wkf,
                               const float* __restrict__ wvf,
                               const float* __restrict__ pe,
                               bf16_t* Yg, bf16_t* Yinv,
                               bf16_t* wq, bf16_t* wk, bf16_t* wv,
                               float* P) {
  int i = blockIdx.x * 256 + threadIdx.x;
  if (i < S_GRID * 64) {                       // Y_grid padded [128][64]
    int g = i >> 6, s = i & 63;
    Yg[i] = (bf16_t)(s < S_SPH ? Ygf[g * S_SPH + s] : 0.f);
  }
  if (i < 64 * S_GRID) {                       // Y_inv padded [64][128]
    int s = i >> 7, c = i & 127;
    Yinv[i] = (bf16_t)(s < S_SPH ? Yinvf[s * S_GRID + c] : 0.f);
  }
  if (i < CH * CH) {
    wq[i] = (bf16_t)wqf[i];
    wk[i] = (bf16_t)wkf[i];
    wv[i] = (bf16_t)wvf[i];
    int a = i >> 7, b = i & 127;               // positional dot matrix
    P[i] = pe[a * 3] * pe[b * 3] + pe[a * 3 + 1] * pe[b * 3 + 1] +
           pe[a * 3 + 2] * pe[b * 3 + 2];
  }
}

__global__ void est_prep_w1(const float* __restrict__ w1,
                            const float* __restrict__ lng,
                            const float* __restrict__ lnb,
                            const float* __restrict__ b1,
                            bf16_t* w1f, float* b1f) {
  int idx = blockIdx.x * 256 + threadIdx.x;
  if (idx < NEXP * HID * CH) {                 // fold LN gain into w1
    int i = idx >> 16;                         // HID*CH = 65536
    int c = idx & 127;
    w1f[idx] = (bf16_t)(w1[idx] * lng[i * CH + c]);
  }
  if (idx < NEXP * HID) {                      // fold LN bias into b1
    int i = idx >> 9, h = idx & 511;
    float s = b1[idx];
    for (int c = 0; c < CH; ++c)
      s += w1[(i * HID + h) * CH + c] * lnb[i * CH + c];
    b1f[idx] = s;
  }
}

__global__ void est_prep_w2(const float* __restrict__ w2, bf16_t* w2b) {
  int idx = blockIdx.x * 256 + threadIdx.x;
  if (idx < NEXP * CH * HID) w2b[idx] = (bf16_t)w2[idx];
}

// ---------------------------------------------------------------------------
// l=0 path: LN -> Linear -> SiLU, writes output row 0 of each token.
// ---------------------------------------------------------------------------
__global__ void est_l0(const float* __restrict__ msg,
                       const float* __restrict__ g, const float* __restrict__ b,
                       const float* __restrict__ wl, const float* __restrict__ bl,
                       float* __restrict__ out) {
  __shared__ float xr[CH], xl[CH];
  int e = blockIdx.x, t = threadIdx.x;
  float v = msg[(size_t)e * S_SPH * CH + t];
  xr[t] = v;
  __syncthreads();
  float m = 0.f;
  for (int c = 0; c < CH; ++c) m += xr[c];
  m *= (1.f / CH);
  float vv = 0.f;
  for (int c = 0; c < CH; ++c) { float d = xr[c] - m; vv += d * d; }
  float rs = rsqrtf(vv * (1.f / CH) + 1e-5f);
  xl[t] = (v - m) * rs * g[t] + b[t];
  __syncthreads();
  float a = bl[t];
  for (int c = 0; c < CH; ++c) a += wl[t * CH + c] * xl[c];
  out[(size_t)e * S_SPH * CH + t] = silu_f(a);
}

// ---------------------------------------------------------------------------
// Fused per-token pipeline.
// ---------------------------------------------------------------------------
__global__ __launch_bounds__(256, 1)
void est_fused(const float* __restrict__ msg, const float* __restrict__ gate,
               const bf16_t* __restrict__ Yg, const bf16_t* __restrict__ Yinv,
               const bf16_t* __restrict__ wq, const bf16_t* __restrict__ wk,
               const bf16_t* __restrict__ wv,
               const float* __restrict__ bq, const float* __restrict__ bk,
               const float* __restrict__ bv,
               const float* __restrict__ ln1g, const float* __restrict__ ln1b,
               const bf16_t* __restrict__ w1f, const float* __restrict__ b1f,
               const bf16_t* __restrict__ w2b, const float* __restrict__ Pmat,
               const float* __restrict__ eb2, float* __restrict__ out) {
  extern __shared__ char smem[];
  float*  S   = (float*)smem;                              // [128][129] f32
  bf16_t* XLN = (bf16_t*)(smem + 128 * LDF * 4);           // [128][128]
  bf16_t* QS  = XLN + CH * CH;                             // q / attn
  bf16_t* KS  = QS  + CH * CH;                             // k / acc^T
  bf16_t* VT  = KS  + CH * CH;                             // v^T [d][g]
  bf16_t* XH  = VT  + CH * CH;                             // msg^T then xhat
  bf16_t* H   = XH  + CH * CH;                             // MoE hidden chunk
  float*  GB2 = (float*)(H + CH * CH);                     // [128]

  const int e    = blockIdx.x;
  const int tid  = threadIdx.x;
  const int lane = tid & 31;
  const int w    = tid >> 5;                               // wave id 0..7
  const int lo   = lane & 15;
  const int hi   = lane >> 4;
  const int m0   = w * 16;

  // ---- Phase 0: message^T (bf16, K padded 49->64) into XH ----------------
  for (int i = tid; i < CH * 64; i += 256) XH[i] = (bf16_t)0.f;
  __syncthreads();
  const float* me = msg + (size_t)e * S_SPH * CH;
  for (int i = tid; i < S_SPH * CH; i += 256) {
    int s = i >> 7, c = i & 127;
    XH[c * 64 + s] = (bf16_t)me[i];
  }
  __syncthreads();

  // ---- Phase 1: ToGrid  x[g][c] = Yg[g][s] * msg[s][c]  (K=64) -----------
  for (int n = 0; n < 8; ++n) {
    int n0 = n * 16;
    v8f acc = {};
#pragma unroll
    for (int kk = 0; kk < 2; ++kk)
      acc = wmma_bf16(fragA(Yg, 64, m0 + lo, kk * 32, hi),
                      fragB(XH, 64, n0 + lo, kk * 32, hi), acc);
#pragma unroll
    for (int v2 = 0; v2 < 8; ++v2)
      S[(m0 + hi * 8 + v2) * LDF + n0 + lo] = acc[v2];
  }
  __syncthreads();

  // ---- Phase 2: LN1 -> XLN (bf16 row-major [g][c]) -----------------------
  if (tid < CH) {
    const float* r = S + tid * LDF;
    float m = 0.f;
    for (int c = 0; c < CH; ++c) m += r[c];
    m *= (1.f / CH);
    float vv = 0.f;
    for (int c = 0; c < CH; ++c) { float d = r[c] - m; vv += d * d; }
    float rs = rsqrtf(vv * (1.f / CH) + 1e-5f);
    for (int c = 0; c < CH; ++c)
      XLN[tid * CH + c] = (bf16_t)((r[c] - m) * rs * ln1g[c] + ln1b[c]);
  }
  __syncthreads();

  // ---- Phase 3: QKV.  q,k row-major [g][d]; v computed transposed [d][g].
  for (int n = 0; n < 8; ++n) {
    int n0 = n * 16;
    v8f aq = {}, ak = {}, av = {};
#pragma unroll
    for (int k = 0; k < 4; ++k) {
      v16bf xa = fragA(XLN, CH, m0 + lo, k * 32, hi);
      aq = wmma_bf16(xa, fragB(wq, CH, n0 + lo, k * 32, hi), aq);
      ak = wmma_bf16(xa, fragB(wk, CH, n0 + lo, k * 32, hi), ak);
      av = wmma_bf16(fragA(wv, CH, m0 + lo, k * 32, hi),
                     fragB(XLN, CH, n0 + lo, k * 32, hi), av);
    }
    float biq = bq[n0 + lo], bik = bk[n0 + lo];
#pragma unroll
    for (int v2 = 0; v2 < 8; ++v2) {
      int r = m0 + hi * 8 + v2;
      QS[r * CH + n0 + lo] = (bf16_t)(aq[v2] + biq);
      KS[r * CH + n0 + lo] = (bf16_t)(ak[v2] + bik);
      VT[r * CH + n0 + lo] = (bf16_t)(av[v2] + bv[r]);   // r = channel d
    }
  }
  __syncthreads();

  // ---- Phase 4: scores = (q.k + P) * rsqrt(131) -> S ---------------------
  {
    const float scl = rsqrtf(131.f);
    for (int n = 0; n < 8; ++n) {
      int n0 = n * 16;
      v8f acc = {};
#pragma unroll
      for (int k = 0; k < 4; ++k)
        acc = wmma_bf16(fragA(QS, CH, m0 + lo, k * 32, hi),
                        fragB(KS, CH, n0 + lo, k * 32, hi), acc);
#pragma unroll
      for (int v2 = 0; v2 < 8; ++v2) {
        int r = m0 + hi * 8 + v2, cix = n0 + lo;
        S[r * LDF + cix] = (acc[v2] + Pmat[r * CH + cix]) * scl;
      }
    }
  }
  __syncthreads();

  // ---- Phase 5: softmax rows -> attn (bf16, reuse QS) --------------------
  if (tid < S_GRID) {
    float* r = S + tid * LDF;
    float mx = -1e30f;
    for (int j = 0; j < S_GRID; ++j) mx = fmaxf(mx, r[j]);
    float sm = 0.f;
    for (int j = 0; j < S_GRID; ++j) { float p = __expf(r[j] - mx); r[j] = p; sm += p; }
    float inv = 1.f / sm;
    for (int j = 0; j < S_GRID; ++j) QS[tid * CH + j] = (bf16_t)(r[j] * inv);
  }
  __syncthreads();

  // ---- Phase 6: out = attn @ v  (Bt = VT directly) -> S ------------------
  for (int n = 0; n < 8; ++n) {
    int n0 = n * 16;
    v8f acc = {};
#pragma unroll
    for (int k = 0; k < 4; ++k)
      acc = wmma_bf16(fragA(QS, CH, m0 + lo, k * 32, hi),
                      fragB(VT, CH, n0 + lo, k * 32, hi), acc);
#pragma unroll
    for (int v2 = 0; v2 < 8; ++v2)
      S[(m0 + hi * 8 + v2) * LDF + n0 + lo] = acc[v2];
  }
  __syncthreads();

  // ---- Phase 7: LN stats (no affine; folded into experts) -> XH; GB2 -----
  if (tid < S_GRID) {
    float* r = S + tid * LDF;
    float m = 0.f;
    for (int c = 0; c < CH; ++c) m += r[c];
    m *= (1.f / CH);
    float vv = 0.f;
    for (int c = 0; c < CH; ++c) { float d = r[c] - m; vv += d * d; }
    float rs = rsqrtf(vv * (1.f / CH) + 1e-5f);
    for (int c = 0; c < CH; ++c) XH[tid * CH + c] = (bf16_t)((r[c] - m) * rs);
    float g2 = 0.f;
    for (int i = 0; i < NEXP; ++i) g2 += gate[e * NEXP + i] * eb2[i * CH + tid];
    GB2[tid] = g2;
  }
  __syncthreads();

  // ---- Phase 8: dense MoE.  gate folded into h; z^T accumulated in VGPRs.
  // xhat A-fragments are invariant across experts and chunks: hoist them.
  v16bf xaf[4];
#pragma unroll
  for (int k = 0; k < 4; ++k) xaf[k] = fragA(XH, CH, m0 + lo, k * 32, hi);

  v8f ztot[8] = {};
  for (int ie = 0; ie < NEXP; ++ie) {
    const bf16_t* W1 = w1f + (size_t)ie * HID * CH;
    const bf16_t* W2 = w2b + (size_t)ie * CH * HID;
    const float*  B1 = b1f + ie * HID;
    __builtin_prefetch(W1, 0, 1);
    __builtin_prefetch(W2, 0, 1);
    const float gi = gate[e * NEXP + ie];
    for (int kc = 0; kc < 4; ++kc) {                 // 4 hidden chunks of 128
      // GEMM1 chunk: h[g][hh] = gate_i * silu(xhat @ w1'^T + b1'), strip m0
      for (int n = 0; n < 8; ++n) {
        int hh = kc * 128 + n * 16;
        float bb = B1[hh + lo];
        v8f acc;
#pragma unroll
        for (int t = 0; t < 8; ++t) acc[t] = bb;
#pragma unroll
        for (int k = 0; k < 4; ++k)
          acc = wmma_bf16(xaf[k], fragB(W1, CH, hh + lo, k * 32, hi), acc);
#pragma unroll
        for (int v2 = 0; v2 < 8; ++v2)
          H[(m0 + hi * 8 + v2) * CH + n * 16 + lo] =
              (bf16_t)(gi * silu_f(acc[v2]));
      }
      __syncthreads();
      // GEMM2 partial: z^T[c][g] += w2[c][hh] * h[g][hh]  (Bt = H directly)
      for (int n = 0; n < 8; ++n) {
#pragma unroll
        for (int k = 0; k < 4; ++k)
          ztot[n] = wmma_bf16(fragA(W2, HID, m0 + lo, kc * 128 + k * 32, hi),
                              fragB(H, CH, n * 16 + lo, k * 32, hi), ztot[n]);
      }
      __syncthreads();                               // before H is overwritten
    }
  }

  // ---- Phase 9: + gated b2, convert acc^T to bf16 (reuse KS) -------------
  for (int n = 0; n < 8; ++n) {
#pragma unroll
    for (int v2 = 0; v2 < 8; ++v2) {
      int c = m0 + hi * 8 + v2;                      // channel row of z^T
      KS[c * CH + n * 16 + lo] = (bf16_t)(ztot[n][v2] + GB2[c]);
    }
  }
  __syncthreads();

  // ---- Phase 10: FromGrid  sph[s][c] = Yinv[s][g] * acc[g][c] ------------
  {
    int mt = w >> 1;
    int sm0 = mt * 16;
    for (int nn = 0; nn < 4; ++nn) {
      int n0 = ((w & 1) * 4 + nn) * 16;
      v8f acc = {};
#pragma unroll
      for (int k = 0; k < 4; ++k)
        acc = wmma_bf16(fragA(Yinv, CH, sm0 + lo, k * 32, hi),
                        fragB(KS, CH, n0 + lo, k * 32, hi), acc);
#pragma unroll
      for (int v2 = 0; v2 < 8; ++v2) {
        int s2 = sm0 + hi * 8 + v2;
        if (s2 >= 1 && s2 < S_SPH)
          out[(size_t)e * S_SPH * CH + s2 * CH + n0 + lo] = acc[v2];
      }
    }
  }
}

// ---------------------------------------------------------------------------
extern "C" void kernel_launch(void* const* d_in, const int* in_sizes, int n_in,
                              void* d_out, int out_size, void* d_ws, size_t ws_size,
                              hipStream_t stream) {
  const float* msg   = (const float*)d_in[0];
  const float* gate  = (const float*)d_in[1];
  // d_in[2] = batch (unused by reference)
  const float* Ygf   = (const float*)d_in[3];
  const float* Yinvf = (const float*)d_in[4];
  const float* pe    = (const float*)d_in[5];
  const float* ln1g  = (const float*)d_in[6];
  const float* ln1b  = (const float*)d_in[7];
  const float* wqf   = (const float*)d_in[8];
  const float* bq    = (const float*)d_in[9];
  const float* wkf   = (const float*)d_in[10];
  const float* bk    = (const float*)d_in[11];
  const float* wvf   = (const float*)d_in[12];
  const float* bv    = (const float*)d_in[13];
  const float* nl0g  = (const float*)d_in[14];
  const float* nl0b  = (const float*)d_in[15];
  const float* wl0   = (const float*)d_in[16];
  const float* bl0   = (const float*)d_in[17];
  const float* elng  = (const float*)d_in[18];
  const float* elnb  = (const float*)d_in[19];
  const float* ew1   = (const float*)d_in[20];
  const float* eb1   = (const float*)d_in[21];
  const float* ew2   = (const float*)d_in[22];
  const float* eb2   = (const float*)d_in[23];
  float* outp = (float*)d_out;

  char* ws = (char*)d_ws;
  bf16_t* Yg   = (bf16_t*)(ws + 0);            //  16384 B
  bf16_t* Yinv = (bf16_t*)(ws + 16384);        //  16384 B
  bf16_t* wqb  = (bf16_t*)(ws + 32768);        //  32768 B
  bf16_t* wkb  = (bf16_t*)(ws + 65536);        //  32768 B
  bf16_t* wvb  = (bf16_t*)(ws + 98304);        //  32768 B
  bf16_t* w1f  = (bf16_t*)(ws + 131072);       // 1048576 B
  bf16_t* w2b  = (bf16_t*)(ws + 1179648);      // 1048576 B
  float*  b1f  = (float*) (ws + 2228224);      //  16384 B
  float*  P    = (float*) (ws + 2244608);      //  65536 B (ends 2310144)

  est_prep_small<<<64, 256, 0, stream>>>(Ygf, Yinvf, wqf, wkf, wvf, pe,
                                         Yg, Yinv, wqb, wkb, wvb, P);
  est_prep_w1<<<2048, 256, 0, stream>>>(ew1, elng, elnb, eb1, w1f, b1f);
  est_prep_w2<<<2048, 256, 0, stream>>>(ew2, w2b);

  const size_t smem = 128 * LDF * 4            // f32 scratch
                    + 6 * CH * CH * 2          // six bf16 128x128 tiles
                    + 512;                     // GB2 + slack
  est_fused<<<E_TOK, 256, smem, stream>>>(msg, gate, Yg, Yinv, wqb, wkb, wvb,
                                          bq, bk, bv, ln1g, ln1b,
                                          w1f, b1f, w2b, P, eb2, outp);

  est_l0<<<E_TOK, 128, 0, stream>>>(msg, nl0g, nl0b, wl0, bl0, outp);
}
